// AttentionRNN_79078937853994
// MI455X (gfx1250) — compile-verified, matrix operand order checked by reading, not run on hardware
//
#include <hip/hip_runtime.h>
#include <hip/hip_bf16.h>

// AttentionRNN on MI455X (gfx1250).
// softmax over a size-1 axis == 1.0 exactly -> attention branch numerically
// dead; output is an LSTM over x[:,:,t] emitting cell states.
// K1: hoisted GEMM [4096,1024]x[1024,256] bf16 WMMA, B panels via TDM
//     (tensor_load_to_lds, double buffered, s_wait_tensorcnt).
// K2: persistent single-WGP recurrence, h double-buffered in LDS.

typedef __attribute__((ext_vector_type(16))) __bf16    v16bf;
typedef __attribute__((ext_vector_type(8)))  float     v8f;
typedef __attribute__((ext_vector_type(4)))  unsigned  v4u;
typedef __attribute__((ext_vector_type(8)))  int       v8i;
typedef __attribute__((ext_vector_type(4)))  int       v4i;

#define B_SZ 128   // batch
#define F_SZ 1024  // features (K of big GEMM)
#define W_SZ 32    // window (timesteps)
#define H_SZ 64    // hidden
#define G4_SZ 256  // 4*H gate width

// 16-bit operand fragment K index (CDNA5 WMMA layout):
// lanes 0-15: K=0..7 (e 0-7), K=16..23 (e 8-15); lanes 16-31: K=8..15, 24..31.
__device__ __forceinline__ int frag_k(int lane, int e) {
  const int klo = (lane & 16) ? 8 : 0;
  return (e < 8) ? (klo + e) : (16 + klo + (e - 8));
}

__device__ __forceinline__ v8f wmma_bf16(v16bf a, v16bf b, v8f c) {
  return __builtin_amdgcn_wmma_f32_16x16x32_bf16(
      false, a, false, b, (short)0, c, false, false);
}

__device__ __forceinline__ float sigmoidf_fast(float x) {
  return 1.0f / (1.0f + __expf(-x));
}

// ---------------------------------------------------------------------------
// TDM 2-D tile load: global (bf16, row stride stride0 elements) -> LDS packed.
// D# per CDNA5 ISA: count=1, type=2 ("image"),
// data_size=1 (2 bytes), no multicast / iterate / pad / gather.
// ---------------------------------------------------------------------------
__device__ __forceinline__ void tdm_load_2d(unsigned lds_addr, const void* gptr,
                                            unsigned tile_d0, unsigned tile_d1,
                                            unsigned tensor_d0, unsigned tensor_d1,
                                            unsigned stride0) {
  const unsigned long long ga = (unsigned long long)gptr;
  v4u g0;
  g0[0] = 1u;                                        // count=1, user mode
  g0[1] = lds_addr;                                  // lds_addr [63:32]
  g0[2] = (unsigned)(ga & 0xffffffffu);              // global_addr lo
  g0[3] = (unsigned)((ga >> 32) & 0x01ffffffu)       // global_addr hi (57b)
          | (2u << 30);                              // type=2
  v8i g1;
  g1[0] = (int)(1u << 16);                           // data_size=1 (2B)
  g1[1] = (int)((tensor_d0 & 0xffffu) << 16);        // tensor_dim0[15:0]@[63:48]
  g1[2] = (int)(((tensor_d0 >> 16) & 0xffffu) |      // tensor_dim0[31:16]
                ((tensor_d1 & 0xffffu) << 16));      // tensor_dim1[15:0]
  g1[3] = (int)(((tensor_d1 >> 16) & 0xffffu) |      // tensor_dim1[31:16]
                ((tile_d0 & 0xffffu) << 16));        // tile_dim0
  g1[4] = (int)(tile_d1 & 0xffffu);                  // tile_dim1; tile_dim2=0
  g1[5] = (int)stride0;                              // tensor_dim0_stride lo32
  g1[6] = 0;                                         // stride0 hi / stride1 lo
  g1[7] = 0;                                         // stride1 hi
  v4i g2; g2[0] = g2[1] = g2[2] = g2[3] = 0;         // 2-D: groups 2/3 unused
  v4i g3; g3[0] = g3[1] = g3[2] = g3[3] = 0;
#if defined(__clang_major__) && __clang_major__ >= 23
  v8i g4; g4[0]=g4[1]=g4[2]=g4[3]=g4[4]=g4[5]=g4[6]=g4[7]=0;
  __builtin_amdgcn_tensor_load_to_lds(g0, g1, g2, g3, g4, 0);
#else
  __builtin_amdgcn_tensor_load_to_lds(g0, g1, g2, g3, 0);
#endif
}

// ---------------------------------------------------------------------------
// Kernel 0: pack Wx [F,4H] f32 -> Wt [4H,F] bf16 ([n][k], rows contiguous in k)
// so the GEMM's B panels are plain 2-D tiles for the TDM. 32 blocks x 256.
// ---------------------------------------------------------------------------
__global__ __launch_bounds__(256) void pack_wt(const float* __restrict__ Wx,
                                               __bf16* __restrict__ Wt) {
  __shared__ __bf16 ldsW[32 * G4_SZ];
  const int tid = threadIdx.x;
  const int kb  = blockIdx.x * 32;
  for (int i = tid; i < 32 * G4_SZ; i += 256) {      // coalesced over n
    const int kk = i >> 8, n = i & 255;
    ldsW[kk * G4_SZ + n] = (__bf16)Wx[(size_t)(kb + kk) * G4_SZ + n];
  }
  __syncthreads();
  for (int i = tid; i < 32 * G4_SZ; i += 256) {      // coalesced over k
    const int n = i >> 5, kk = i & 31;
    Wt[(size_t)n * F_SZ + kb + kk] = ldsW[kk * G4_SZ + n];
  }
}

// ---------------------------------------------------------------------------
// Kernel 1: G[b*32+t, g] = sum_f x[b,f,t]*Wx[f,g] + b_lstm[g]
// One block per batch b. A panel (32x1024) converted to bf16 LDS once.
// B panels (256x32 bf16) pulled by TDM, double buffered, overlapped with WMMA.
// ---------------------------------------------------------------------------
__global__ __launch_bounds__(256) void gates_x_gemm(
    const float*  __restrict__ x,       // [B, F, W]
    const __bf16* __restrict__ Wt,      // [4H, F] bf16
    const float*  __restrict__ b_lstm,  // [4H]
    float* __restrict__ G) {            // [B*W, 4H]
  __shared__ __bf16 ldsA[32 * F_SZ];    // full A panel [t][f], 64 KB
  __shared__ __bf16 ldsB[2][G4_SZ * 32];// B double buffer [n][k], 2 x 16 KB

  const int tid  = threadIdx.x;
  const int lane = tid & 31;
  const int wave = tid >> 5;
  const int b    = blockIdx.x;
  const int Mt   = wave >> 2;           // 0..1
  const int nset = wave & 3;            // N-tiles nset*4 .. +3
  const int ln15 = lane & 15;

  // Phase 1: convert-transpose x panel -> ldsA (float4 loads, no waits chain)
  const size_t xb = (size_t)b * F_SZ * W_SZ;
  for (int c = tid; c < (F_SZ * W_SZ) / 4; c += 256) {
    const int f  = c >> 3;
    const int t0 = (c & 7) * 4;
    const float4 v = *(const float4*)&x[xb + (size_t)f * W_SZ + t0];
    ldsA[(t0 + 0) * F_SZ + f] = (__bf16)v.x;
    ldsA[(t0 + 1) * F_SZ + f] = (__bf16)v.y;
    ldsA[(t0 + 2) * F_SZ + f] = (__bf16)v.z;
    ldsA[(t0 + 3) * F_SZ + f] = (__bf16)v.w;
  }

  // Kick off first B panel DMA (one wave owns the TDM + TENSORcnt)
  if (wave == 0)
    tdm_load_2d((unsigned)(size_t)&ldsB[0][0], Wt,
                /*tile*/32, G4_SZ, /*tensor*/F_SZ, G4_SZ, /*stride0*/F_SZ);

  v8f acc[4];
#pragma unroll
  for (int i = 0; i < 4; ++i)
#pragma unroll
    for (int s = 0; s < 8; ++s) acc[i][s] = 0.0f;

  const int r = Mt * 16 + ln15;

  for (int it = 0; it < F_SZ / 32; ++it) {
    if (wave == 0) __builtin_amdgcn_s_wait_tensorcnt(0);
    __syncthreads();  // B panel (and, on it==0, the A panel) visible to all

    // Overlap: start next B panel DMA while this one is consumed
    if (wave == 0 && it + 1 < F_SZ / 32)
      tdm_load_2d((unsigned)(size_t)&ldsB[(it + 1) & 1][0],
                  Wt + (it + 1) * 32, 32, G4_SZ, F_SZ, G4_SZ, F_SZ);

    const __bf16* bb = &ldsB[it & 1][0];
    const int kb = it * 32;

    v16bf a;
#pragma unroll
    for (int e = 0; e < 16; ++e) a[e] = ldsA[r * F_SZ + kb + frag_k(lane, e)];

#pragma unroll
    for (int i = 0; i < 4; ++i) {
      const int n = (nset * 4 + i) * 16 + ln15;
      v16bf bf;
#pragma unroll
      for (int e = 0; e < 16; ++e) bf[e] = bb[n * 32 + frag_k(lane, e)];
      acc[i] = wmma_bf16(a, bf, acc[i]);
    }
    // next-iteration barrier doubles as "done reading this buffer"
  }

  // Store C (+bias). C layout: lanes 0-15: M=slot, lanes 16-31: M=8+slot.
#pragma unroll
  for (int i = 0; i < 4; ++i) {
    const int col  = (nset * 4 + i) * 16 + ln15;
    const float bias = b_lstm[col];
#pragma unroll
    for (int s = 0; s < 8; ++s) {
      const int row = b * W_SZ + Mt * 16 + ((lane & 16) ? 8 + s : s);
      G[(size_t)row * G4_SZ + col] = acc[i][s] + bias;
    }
  }
}

// ---------------------------------------------------------------------------
// Kernel 2: persistent single-workgroup LSTM scan, t = 0..31.
// Wave (Mp=w>>2, j=w&3) owns N-tiles {j, j+4, j+8, j+12}: one per gate at the
// same hidden columns -> gate fusion is wave-local. c in accumulator regs;
// h double-buffered in LDS (bf16). Wh fragments register-resident, staged
// through the (not-yet-live) ldsH space with coalesced loads.
// ---------------------------------------------------------------------------
__global__ __launch_bounds__(256) void lstm_scan(
    const float* __restrict__ Wh,   // [H, 4H]
    const float* __restrict__ G,    // [B*W, 4H]
    float* __restrict__ out) {      // [B, W, H]
  __shared__ __bf16 ldsH[2][B_SZ * H_SZ];  // 2 x 16 KB

  const int tid  = threadIdx.x;
  const int lane = tid & 31;
  const int wave = tid >> 5;
  const int j    = wave & 3;
  const int Mp   = wave >> 2;
  const int ln15 = lane & 15;
  const int hcol = j * 16 + ln15;

  // Stage Wh (bf16, [k][n], 32 KB == entire ldsH) with coalesced loads,
  // gather register-resident B fragments, then reclaim the LDS for h.
  __bf16* ldsFlat = &ldsH[0][0];
  for (int i = tid; i < H_SZ * G4_SZ; i += 256)
    ldsFlat[i] = (__bf16)Wh[i];
  __syncthreads();

  v16bf bfrag[4][2];
#pragma unroll
  for (int gi = 0; gi < 4; ++gi) {
    const int n = gi * 64 + hcol;
#pragma unroll
    for (int kc = 0; kc < 2; ++kc)
#pragma unroll
      for (int e = 0; e < 16; ++e)
        bfrag[gi][kc][e] = ldsFlat[(kc * 32 + frag_k(lane, e)) * G4_SZ + n];
  }
  __syncthreads();

  // h buffer 0 <- 0 (buffer 1 is fully overwritten at t=0, no need to zero)
  for (int i = tid; i < B_SZ * H_SZ; i += 256) ldsH[0][i] = (__bf16)0.0f;

  float creg[4][8];
#pragma unroll
  for (int mi = 0; mi < 4; ++mi)
#pragma unroll
    for (int s = 0; s < 8; ++s) creg[mi][s] = 0.0f;

  __syncthreads();

  for (int t = 0; t < W_SZ; ++t) {
    const __bf16* __restrict__ hbuf = ldsH[t & 1];
    __bf16* __restrict__ hnew       = ldsH[(t + 1) & 1];

#pragma unroll
    for (int mi = 0; mi < 4; ++mi) {
      const int Mt = Mp * 4 + mi;
      const int r  = Mt * 16 + ln15;

      v16bf a0, a1;
#pragma unroll
      for (int e = 0; e < 16; ++e) {
        a0[e] = hbuf[r * H_SZ + frag_k(lane, e)];
        a1[e] = hbuf[r * H_SZ + 32 + frag_k(lane, e)];
      }

      v8f acc[4];
#pragma unroll
      for (int gi = 0; gi < 4; ++gi) {
        const int col = gi * 64 + hcol;
#pragma unroll
        for (int s = 0; s < 8; ++s) {
          const int row = Mt * 16 + ((lane & 16) ? 8 + s : s);
          acc[gi][s] = G[((size_t)row * W_SZ + t) * G4_SZ + col];
        }
        acc[gi] = wmma_bf16(a0, bfrag[gi][0], acc[gi]);
        acc[gi] = wmma_bf16(a1, bfrag[gi][1], acc[gi]);
      }

#pragma unroll
      for (int s = 0; s < 8; ++s) {
        const int row = Mt * 16 + ((lane & 16) ? 8 + s : s);
        const float ig = sigmoidf_fast(acc[0][s]);
        const float fg = sigmoidf_fast(acc[1][s]);
        const float gg = tanhf(acc[2][s]);
        const float og = sigmoidf_fast(acc[3][s]);
        const float c  = fg * creg[mi][s] + ig * gg;
        const float h  = og * tanhf(c);
        creg[mi][s] = c;
        out[((size_t)row * W_SZ + t) * H_SZ + hcol] = c;  // cell state
        hnew[row * H_SZ + hcol] = (__bf16)h;
      }
    }
    __syncthreads();
  }
}

// ---------------------------------------------------------------------------
extern "C" void kernel_launch(void* const* d_in, const int* in_sizes, int n_in,
                              void* d_out, int out_size, void* d_ws, size_t ws_size,
                              hipStream_t stream) {
  // 0:x 1:W_state 2:b_state 3:W_in 4:w_attn 5:b_attn 6:Wx 7:Wh 8:b_lstm
  // Inputs 1-5 are numerically dead (softmax over size-1 axis == 1 exactly).
  const float* x      = (const float*)d_in[0];
  const float* Wx     = (const float*)d_in[6];
  const float* Wh     = (const float*)d_in[7];
  const float* b_lstm = (const float*)d_in[8];

  float*  G  = (float*)d_ws;                                  // 4 MB
  __bf16* Wt = (__bf16*)((char*)d_ws + (size_t)B_SZ * W_SZ * G4_SZ * 4);  // 512 KB

  float* out = (float*)d_out;

  pack_wt<<<dim3(F_SZ / 32), dim3(256), 0, stream>>>(Wx, Wt);
  gates_x_gemm<<<dim3(B_SZ), dim3(256), 0, stream>>>(x, Wt, b_lstm, G);
  lstm_scan<<<dim3(1), dim3(256), 0, stream>>>(Wh, G, out);
}
